// MambaAttn_14190571946216
// MI455X (gfx1250) — compile-verified, matrix operand order checked by reading, not run on hardware
//
#include <hip/hip_runtime.h>
#include <hip/hip_bf16.h>

typedef __attribute__((ext_vector_type(16))) _Float16 v16h;
typedef __attribute__((ext_vector_type(8)))  _Float16 v8h;
typedef __attribute__((ext_vector_type(8)))  float    v8f;

#define TOK 51200   // 8 * 80 * 80 tokens (same count for global and window order)

__device__ __forceinline__ float silu_f(float x) { return x / (1.0f + __expf(-x)); }

// ---------------------------------------------------------------------------
// WMMA GEMM (K fixed at 128):  O[M, ncols] = A[M,128] @ W[ncols,128]^T * scale
// Block = 128 threads (4 waves) -> one 16x64 C panel.
//  * A strip (16x128) and B strip (64x128, zero-padded past ncols) staged to
//    LDS as f16 with coalesced global loads; OOB masking happens at staging so
//    the WMMA loop is branch-free.
//  * LDS rows padded to 136 halfs (68 dwords) -> fragment reads are
//    conflict-free 16B ds loads (bank = 4*lane).
//  * Fragment packing per ISA 7.12.2: lane m=lane%16; k runs
//    [k0+half*8, +8) and [k0+16+half*8, +8) -> two v8h LDS loads + shuffle.
//  * k-loop fully unrolled -> 4 back-to-back v_wmma_f32_16x16x32_f16.
// ---------------------------------------------------------------------------
__global__ __launch_bounds__(128) void gemm_wmma_f16(
    const float* __restrict__ A, int lda,
    const float* __restrict__ W, int ldw,
    float* __restrict__ O, int ldo,
    int ncols, float scale)
{
  __shared__ _Float16 lA[16][136];
  __shared__ _Float16 lB[64][136];

  const int tid     = threadIdx.x;
  const int wave    = tid >> 5;
  const int lane    = tid & 31;
  const int half_id = lane >> 4;
  const int lid     = lane & 15;
  const int rowbase = blockIdx.x * 16;
  const int colbase = blockIdx.y * 64;

  // ---- stage A: each thread converts 16 consecutive floats of one row
  {
    const int r  = tid >> 3;          // 0..15
    const int kc = (tid & 7) << 4;    // 0,16,...,112
    const float* src = A + (size_t)(rowbase + r) * lda + kc;
    v8h lo = {}, hi = {};
#pragma unroll
    for (int e = 0; e < 8; ++e) {
      lo[e] = (_Float16)src[e];
      hi[e] = (_Float16)src[8 + e];
    }
    *(v8h*)&lA[r][kc]     = lo;
    *(v8h*)&lA[r][kc + 8] = hi;
  }
  // ---- stage B: 64 output columns x 128 k, zeros beyond ncols
#pragma unroll
  for (int rep = 0; rep < 4; ++rep) {
    const int u  = tid + rep * 128;   // 0..511
    const int cr = u >> 3;            // 0..63
    const int kc = (u & 7) << 4;
    const int gc = colbase + cr;
    v8h lo = {}, hi = {};
    if (gc < ncols) {
      const float* src = W + (size_t)gc * ldw + kc;
#pragma unroll
      for (int e = 0; e < 8; ++e) {
        lo[e] = (_Float16)src[e];
        hi[e] = (_Float16)src[8 + e];
      }
    }
    *(v8h*)&lB[cr][kc]     = lo;
    *(v8h*)&lB[cr][kc + 8] = hi;
  }
  __syncthreads();

  const int crow = wave * 16 + lid;   // lB row (output column) for this lane
  v8f acc = {};
#pragma unroll
  for (int k0 = 0; k0 < 128; k0 += 32) {
    const int ko = k0 + half_id * 8;
    const v8h a0 = *(const v8h*)&lA[lid][ko];
    const v8h a1 = *(const v8h*)&lA[lid][ko + 16];
    const v8h b0 = *(const v8h*)&lB[crow][ko];
    const v8h b1 = *(const v8h*)&lB[crow][ko + 16];
    const v16h a = __builtin_shufflevector(a0, a1, 0, 1, 2, 3, 4, 5, 6, 7,
                                           8, 9, 10, 11, 12, 13, 14, 15);
    const v16h b = __builtin_shufflevector(b0, b1, 0, 1, 2, 3, 4, 5, 6, 7,
                                           8, 9, 10, 11, 12, 13, 14, 15);
    acc = __builtin_amdgcn_wmma_f32_16x16x32_f16(false, a, false, b, (short)0, acc,
                                                 false, false);
  }
#pragma unroll
  for (int v = 0; v < 8; ++v) {
    const int m = rowbase + half_id * 8 + v;
    const int n = colbase + wave * 16 + lid;
    if (n < ncols) O[(size_t)m * ldo + n] = acc[v] * scale;
  }
}

// ---------------------------------------------------------------------------
// Depthwise causal conv1d (K=4) + SiLU. `rev` processes each sequence reversed
// (bwd branch); output stored in logical (possibly reversed) order.
// ---------------------------------------------------------------------------
__global__ void conv1d_silu(const float* __restrict__ xin, int rstride,
                            const float* __restrict__ cw, const float* __restrict__ cb,
                            float* __restrict__ xc, int L, int rev, int total)
{
  int idx = blockIdx.x * blockDim.x + threadIdx.x;
  if (idx >= total) return;
  const int d = idx & 127;
  const int t = idx >> 7;
  const int s = t / L;
  const int l = t - s * L;
  float acc = cb[d];
#pragma unroll
  for (int j = 0; j < 4; ++j) {
    const int ls = l - 3 + j;
    if (ls >= 0) {
      const int li = rev ? (L - 1 - ls) : ls;
      acc += cw[d * 4 + j] * xin[(size_t)(s * L + li) * rstride + d];
    }
  }
  xc[idx] = silu_f(acc);
}

// dt = softplus(xdbl[:, :8] @ dt_proj_w^T + dt_proj_b)
__global__ void dt_softplus(const float* __restrict__ xdbl,
                            const float* __restrict__ dtw, const float* __restrict__ dtb,
                            float* __restrict__ dt, int total)
{
  int idx = blockIdx.x * blockDim.x + threadIdx.x;
  if (idx >= total) return;
  const int d = idx & 127;
  const int t = idx >> 7;
  float acc = dtb[d];
#pragma unroll
  for (int k = 0; k < 8; ++k) acc += xdbl[(size_t)t * 32 + k] * dtw[d * 8 + k];
  dt[idx] = (acc > 20.f) ? acc : log1pf(__expf(acc));
}

// ---------------------------------------------------------------------------
// Selective scan: one thread per (sequence, channel), N states in registers.
// xdbl row layout (ld 32): [0..7]=dt inputs, [8..8+N)=B, [8+N..8+2N)=C.
// ---------------------------------------------------------------------------
template <int N>
__global__ __launch_bounds__(128) void selective_scan(
    const float* __restrict__ xc, const float* __restrict__ dtb,
    const float* __restrict__ xdbl, const float* __restrict__ alog,
    const float* __restrict__ Dw, float* __restrict__ y, int L)
{
  const int s = blockIdx.x;
  const int d = threadIdx.x;
  float a[N], h[N];
#pragma unroll
  for (int n = 0; n < N; ++n) { a[n] = -__expf(alog[d * N + n]); h[n] = 0.f; }
  const float Dd = Dw[d];
  for (int l = 0; l < L; ++l) {
    const int t = s * L + l;
    const float dtv = dtb[(size_t)t * 128 + d];
    const float u   = xc[(size_t)t * 128 + d];
    const float du  = dtv * u;
    float yv = 0.f;
#pragma unroll
    for (int n = 0; n < N; ++n) {
      const float Bt = xdbl[(size_t)t * 32 + 8 + n];
      const float Ct = xdbl[(size_t)t * 32 + 8 + N + n];
      h[n] = __expf(dtv * a[n]) * h[n] + du * Bt;
      yv += h[n] * Ct;
    }
    y[(size_t)t * 128 + d] = yv + u * Dd;
  }
}

// (yf[l] + yb[L-1-l]) * silu(z[l]), then RMSNorm over d=128 with norm_w.
__global__ __launch_bounds__(128) void combine_rmsnorm(
    const float* __restrict__ yf, const float* __restrict__ yb,
    const float* __restrict__ xz, const float* __restrict__ nw,
    float* __restrict__ out, int L)
{
  const int t = blockIdx.x;
  const int d = threadIdx.x;
  const int s = t / L;
  const int l = t - s * L;
  const int tr = s * L + (L - 1 - l);
  const float zv = xz[(size_t)t * 256 + 128 + d];
  const float v = (yf[(size_t)t * 128 + d] + yb[(size_t)tr * 128 + d]) * silu_f(zv);
  __shared__ float red[128];
  red[d] = v * v;
  __syncthreads();
  for (int off = 64; off > 0; off >>= 1) {
    if (d < off) red[d] += red[d + off];
    __syncthreads();
  }
  const float r = rsqrtf(red[0] * (1.0f / 128.0f) + 1e-5f);
  out[(size_t)t * 128 + d] = v * r * nw[d];
}

// spatial (b,80,80,c) -> window order (b,16,16,5*5,c)
__global__ void gather_windows(const float* __restrict__ x, float* __restrict__ xl, int total)
{
  int idx = blockIdx.x * blockDim.x + threadIdx.x;
  if (idx >= total) return;
  const int c = idx & 127;
  const int tw = idx >> 7;
  const int b = tw / 6400;
  int r = tw - b * 6400;
  const int i = r / 400;  r -= i * 400;   // 16 * 25
  const int j = r / 25;   const int p = r - j * 25;
  const int pi = p / 5, pj = p - pi * 5;
  const int hh = i * 5 + pi, ww = j * 5 + pj;
  xl[idx] = x[(((size_t)(b * 80 + hh)) * 80 + ww) * 128 + c];
}

// pre[spatial] = xg[spatial] + xl[window-order]
__global__ void scatter_add(const float* __restrict__ xg, const float* __restrict__ xl,
                            float* __restrict__ pre, int total)
{
  int idx = blockIdx.x * blockDim.x + threadIdx.x;
  if (idx >= total) return;
  const int c = idx & 127;
  const int ts = idx >> 7;
  const int b = ts / 6400;
  const int r = ts - b * 6400;
  const int hh = r / 80, ww = r - hh * 80;
  const int i = hh / 5, pi = hh - i * 5;
  const int j = ww / 5, pj = ww - j * 5;
  const int tw = ((b * 16 + i) * 16 + j) * 25 + pi * 5 + pj;
  pre[idx] = xg[idx] + xl[(size_t)tw * 128 + c];
}

extern "C" void kernel_launch(void* const* d_in, const int* in_sizes, int n_in,
                              void* d_out, int out_size, void* d_ws, size_t ws_size,
                              hipStream_t stream)
{
  (void)in_sizes; (void)n_in; (void)out_size; (void)ws_size;
  // Depth-first, insertion-order flattening of setup_inputs():
  const float* x      = (const float*)d_in[0];
  const float* g_in   = (const float*)d_in[1];
  const float* gf_cw  = (const float*)d_in[2];
  const float* gf_cb  = (const float*)d_in[3];
  const float* gf_xp  = (const float*)d_in[4];
  const float* gf_dtw = (const float*)d_in[5];
  const float* gf_dtb = (const float*)d_in[6];
  const float* gf_al  = (const float*)d_in[7];
  const float* gf_D   = (const float*)d_in[8];
  const float* gb_cw  = (const float*)d_in[9];
  const float* gb_cb  = (const float*)d_in[10];
  const float* gb_xp  = (const float*)d_in[11];
  const float* gb_dtw = (const float*)d_in[12];
  const float* gb_dtb = (const float*)d_in[13];
  const float* gb_al  = (const float*)d_in[14];
  const float* gb_D   = (const float*)d_in[15];
  const float* g_nw   = (const float*)d_in[16];
  const float* g_op   = (const float*)d_in[17];
  const float* l_in   = (const float*)d_in[18];
  const float* lf_cw  = (const float*)d_in[19];
  const float* lf_cb  = (const float*)d_in[20];
  const float* lf_xp  = (const float*)d_in[21];
  const float* lf_dtw = (const float*)d_in[22];
  const float* lf_dtb = (const float*)d_in[23];
  const float* lf_al  = (const float*)d_in[24];
  const float* lf_D   = (const float*)d_in[25];
  const float* lb_cw  = (const float*)d_in[26];
  const float* lb_cb  = (const float*)d_in[27];
  const float* lb_xp  = (const float*)d_in[28];
  const float* lb_dtw = (const float*)d_in[29];
  const float* lb_dtb = (const float*)d_in[30];
  const float* lb_al  = (const float*)d_in[31];
  const float* lb_D   = (const float*)d_in[32];
  const float* l_nw   = (const float*)d_in[33];
  const float* l_op   = (const float*)d_in[34];
  const float* w_out  = (const float*)d_in[35];

  float* ws = (float*)d_ws;
  const size_t T = TOK;
  float* XZ = ws;               // T x 256 (xa | z)
  float* F0 = XZ + T * 256;     // six T x 128 rotating buffers
  float* F1 = F0 + T * 128;
  float* F2 = F1 + T * 128;
  float* F3 = F2 + T * 128;
  float* F4 = F3 + T * 128;
  float* F5 = F4 + T * 128;
  float* S0 = F5 + T * 128;     // two T x 32 xdbl buffers
  float* S1 = S0 + T * 32;

  const int total = (int)(T * 128);
  const int eb = 256, eg = (total + eb - 1) / eb;

  auto gemm = [&](const float* A, const float* W, float* O, int ldo,
                  int ncols, float scale) {
    dim3 grid((unsigned)(T / 16), (unsigned)((ncols + 63) / 64));
    gemm_wmma_f16<<<grid, 128, 0, stream>>>(A, 128, W, 128, O, ldo, ncols, scale);
  };

  // ---------------- global mamba: 8 sequences of L=6400, N=8 ----------------
  gemm(x, g_in, XZ, 256, 256, 1.0f);                                     // xz
  conv1d_silu<<<eg, eb, 0, stream>>>(XZ, 256, gf_cw, gf_cb, F0, 6400, 0, total);
  conv1d_silu<<<eg, eb, 0, stream>>>(XZ, 256, gb_cw, gb_cb, F1, 6400, 1, total);
  gemm(F0, gf_xp, S0, 32, 24, 1.0f);                                     // xdbl fwd
  gemm(F1, gb_xp, S1, 32, 24, 1.0f);                                     // xdbl bwd
  dt_softplus<<<eg, eb, 0, stream>>>(S0, gf_dtw, gf_dtb, F2, total);
  dt_softplus<<<eg, eb, 0, stream>>>(S1, gb_dtw, gb_dtb, F3, total);
  selective_scan<8><<<8, 128, 0, stream>>>(F0, F2, S0, gf_al, gf_D, F4, 6400);
  selective_scan<8><<<8, 128, 0, stream>>>(F1, F3, S1, gb_al, gb_D, F5, 6400);
  combine_rmsnorm<<<(int)T, 128, 0, stream>>>(F4, F5, XZ, g_nw, F0, 6400);
  gemm(F0, g_op, F1, 128, 128, 0.5f);                                    // F1 = xg

  // ---------------- local mamba: 2048 sequences of L=25, N=2 ----------------
  gather_windows<<<eg, eb, 0, stream>>>(x, F2, total);
  gemm(F2, l_in, XZ, 256, 256, 1.0f);
  conv1d_silu<<<eg, eb, 0, stream>>>(XZ, 256, lf_cw, lf_cb, F3, 25, 0, total);
  conv1d_silu<<<eg, eb, 0, stream>>>(XZ, 256, lb_cw, lb_cb, F4, 25, 1, total);
  gemm(F3, lf_xp, S0, 32, 12, 1.0f);
  gemm(F4, lb_xp, S1, 32, 12, 1.0f);
  dt_softplus<<<eg, eb, 0, stream>>>(S0, lf_dtw, lf_dtb, F5, total);
  dt_softplus<<<eg, eb, 0, stream>>>(S1, lb_dtw, lb_dtb, F0, total);
  selective_scan<2><<<2048, 128, 0, stream>>>(F3, F5, S0, lf_al, lf_D, F2, 25);
  selective_scan<2><<<2048, 128, 0, stream>>>(F4, F0, S1, lb_al, lb_D, F3, 25);
  combine_rmsnorm<<<(int)T, 128, 0, stream>>>(F2, F3, XZ, l_nw, F4, 25);
  gemm(F4, l_op, F5, 128, 128, 0.5f);                                    // F5 = xl

  // ---------------- merge + final projection ----------------
  scatter_add<<<eg, eb, 0, stream>>>(F1, F5, F0, total);
  gemm(F0, w_out, (float*)d_out, 128, 128, 1.0f);
}
